// StackedLSTM_46325517254843
// MI455X (gfx1250) — compile-verified, matrix operand order checked by reading, not run on hardware
//
#include <hip/hip_runtime.h>
#include <hip/hip_bf16.h>
#include <cstdint>

// Stacked RNN (LayerNormLSTM -> PeepholeLSTM -> IndRNN) for MI455X (gfx1250).
// B=64, T=512, I=256, H=512. Strategy:
//  - All big time-parallel projections via v_wmma_f32_16x16x32_f16 (f16 in, f32 acc).
//  - Recurrences: 4 persistent WGs (16 batch rows each = one WMMA M-tile),
//    h/c/gates/gi-tile live in LDS (CDNA5 320KB/WGP), w_hh f16 stays in 192MB L2.
//  - Per-step gi tile (16x2048 f32 = 128KB) prefetched into LDS overlapped with
//    the WMMA phase: rec1 uses GLOBAL_LOAD_ASYNC_TO_LDS_B128 (ASYNCcnt),
//    rec2 uses the Tensor Data Mover (TENSOR_LOAD_TO_LDS, TENSORcnt).
//  - IndRNN scan: elementwise, memory bound.

#define B_   64
#define T_   512
#define I_   256
#define H_   512
#define G4_  2048
#define EPS_ 1e-5f

typedef __attribute__((ext_vector_type(16))) _Float16 v16h;
typedef __attribute__((ext_vector_type(8)))  _Float16 v8h;
typedef __attribute__((ext_vector_type(8)))  float    v8f;
typedef __attribute__((ext_vector_type(4)))  unsigned int u32x4;
typedef __attribute__((ext_vector_type(8)))  int i32x8;
typedef __attribute__((ext_vector_type(4)))  int i32x4;
typedef _Float16 h16;

__device__ __forceinline__ v8f wmma32(v16h a, v16h b, v8f c) {
  // D = A(16x32 f16) x B(32x16 f16) + C(16x16 f32)
  return __builtin_amdgcn_wmma_f32_16x16x32_f16(
      /*neg_a=*/false, a, /*neg_b=*/false, b,
      /*c_mod=*/(short)0, c, /*reuse_a=*/false, /*reuse_b=*/false);
}

// Load one 16x32 f16 fragment (A-layout; also used for B by loading w rows,
// since B[k,n] = w[n,k]). Per ISA 7.12.2.
__device__ __forceinline__ v16h ldfrag(const h16* __restrict__ p, int ld) {
  const int lane = threadIdx.x & 31;
  const h16* q = p + (size_t)(lane & 15) * ld + ((lane >> 4) << 3);
  v8h lo = *(const v8h*)(q);
  v8h hi = *(const v8h*)(q + 16);
  v16h f;
#pragma unroll
  for (int i = 0; i < 8; ++i) { f[i] = lo[i]; f[i + 8] = hi[i]; }
  return f;
}

__device__ __forceinline__ float sigm(float x) { return 1.f / (1.f + __expf(-x)); }

// ---------------------------------------------------------------- convert ---
__global__ __launch_bounds__(256)
void f32_to_f16_kernel(const float* __restrict__ in, h16* __restrict__ out, size_t n) {
  size_t i = (size_t)blockIdx.x * blockDim.x + threadIdx.x;
  size_t stride = (size_t)gridDim.x * blockDim.x;
  for (; i < n; i += stride) out[i] = (h16)in[i];
}

// -------------------------------------------------------------------- GEMM ---
// C[m,n] = sum_k A[m,k] * W[n,k] + bias[n].  WG tile 128x128, 8 waves (4Mx2N),
// wave tile 32x64 -> 2x4 WMMA accumulators.
__global__ __launch_bounds__(256)
void gemm_bias_f16(const h16* __restrict__ A, const h16* __restrict__ W,
                   const float* __restrict__ bias, float* __restrict__ C,
                   int M, int N, int K) {
  const int wid   = threadIdx.x >> 5;
  const int lane  = threadIdx.x & 31;
  const int mBase = blockIdx.y * 128 + (wid >> 1) * 32;
  const int nBase = blockIdx.x * 128 + (wid & 1) * 64;

  const v8f z8 = {0.f, 0.f, 0.f, 0.f, 0.f, 0.f, 0.f, 0.f};
  v8f acc[2][4];
#pragma unroll
  for (int mt = 0; mt < 2; ++mt)
#pragma unroll
    for (int nt = 0; nt < 4; ++nt) acc[mt][nt] = z8;

  for (int k = 0; k < K; k += 32) {
    __builtin_prefetch(A + (size_t)mBase * K + k + 128, 0, 3);        // global_prefetch_b8
    __builtin_prefetch(W + (size_t)nBase * K + k + 128, 0, 3);
    v16h a[2], b[4];
#pragma unroll
    for (int mt = 0; mt < 2; ++mt)
      a[mt] = ldfrag(A + (size_t)(mBase + mt * 16) * K + k, K);
#pragma unroll
    for (int nt = 0; nt < 4; ++nt)
      b[nt] = ldfrag(W + (size_t)(nBase + nt * 16) * K + k, K);
#pragma unroll
    for (int mt = 0; mt < 2; ++mt)
#pragma unroll
      for (int nt = 0; nt < 4; ++nt)
        acc[mt][nt] = wmma32(a[mt], b[nt], acc[mt][nt]);
  }

  const int nOff = lane & 15;
  const int mTop = (lane >> 4) << 3;   // 0 or 8
#pragma unroll
  for (int mt = 0; mt < 2; ++mt)
#pragma unroll
    for (int nt = 0; nt < 4; ++nt) {
      const int n  = nBase + nt * 16 + nOff;
      const float bv = bias[n];
      const int m0 = mBase + mt * 16 + mTop;
#pragma unroll
      for (int v = 0; v < 8; ++v)
        C[(size_t)(m0 + v) * N + n] = acc[mt][nt][v] + bv;
    }
}

// --------------------------------------------------------- row LayerNorm ----
__global__ __launch_bounds__(256)
void ln_rows_kernel(float* __restrict__ X, const float* __restrict__ g,
                    const float* __restrict__ be, int width) {
  float* x = X + (size_t)blockIdx.x * width;
  float s = 0.f, s2 = 0.f;
  for (int i = threadIdx.x; i < width; i += 256) { float v = x[i]; s += v; s2 += v * v; }
#pragma unroll
  for (int o = 16; o; o >>= 1) { s += __shfl_xor(s, o, 32); s2 += __shfl_xor(s2, o, 32); }
  __shared__ float rs[8], rq[8];
  if ((threadIdx.x & 31) == 0) { rs[threadIdx.x >> 5] = s; rq[threadIdx.x >> 5] = s2; }
  __syncthreads();
  s = 0.f; s2 = 0.f;
#pragma unroll
  for (int w = 0; w < 8; ++w) { s += rs[w]; s2 += rq[w]; }
  const float mean = s / width;
  const float rstd = rsqrtf(s2 / width - mean * mean + EPS_);
  for (int i = threadIdx.x; i < width; i += 256)
    x[i] = (x[i] - mean) * rstd * g[i] + be[i];
}

// --------------------------------------------- layer 1: LayerNorm LSTM scan --
// grid 4 (16 batch rows per WG), block 512 (16 waves). LDS byte layout:
//   [0)        sh   : [16][H]  f16   (16384)
//   [16384)    sc   : [16][H]  f32   (32768)
//   [49152)    sgh  : [16][4H] f32   (131072)
//   [180224)   stats: 64 f32         (256)
//   [180480)   sgi  : [16][4H] f32   (131072)   <- async-staged gi(t) tile
#define SGI_OFF1 180480u
__global__ __launch_bounds__(512)
void rec_lnlstm_kernel(const float* __restrict__ gi,   // [B,T,4H], already LN'd
                       const h16* __restrict__ whh,    // [4H,H] f16
                       const float* __restrict__ b_hh,
                       const float* __restrict__ g_hh, const float* __restrict__ be_hh,
                       const float* __restrict__ g_c,  const float* __restrict__ be_c,
                       h16* __restrict__ outh) {       // [B,T,H] f16
  extern __shared__ unsigned char smem[];
  h16*   sh    = (h16*)smem;
  float* sc    = (float*)(smem + (size_t)16 * H_ * sizeof(h16));
  float* sgh   = sc + 16 * H_;
  float* smean = sgh + 16 * G4_;
  float* srstd = smean + 16;
  float* scm   = srstd + 16;
  float* scr   = scm + 16;
  float* sgi   = (float*)(smem + SGI_OFF1);

  const int b0 = blockIdx.x * 16;
  const int tid = threadIdx.x, wid = tid >> 5, lane = tid & 31;
  const int nSlab = wid * 128;
  const v8f z8 = {0.f, 0.f, 0.f, 0.f, 0.f, 0.f, 0.f, 0.f};

  for (int i = tid; i < 16 * H_; i += 512) { sh[i] = (h16)0.f; sc[i] = 0.f; }
  __syncthreads();

  for (int t = 0; t < T_; ++t) {
    // ---- async-stage this step's gi tile into LDS (overlaps the WMMA phase)
#pragma unroll
    for (int p = 0; p < 16; ++p) {
      const int flat4 = (p * 512 + tid) * 4;          // float index in 16x2048 tile
      const int r = flat4 >> 11, col = flat4 & (G4_ - 1);
      const unsigned long long ga = (unsigned long long)(uintptr_t)
          (gi + ((size_t)(b0 + r) * T_ + t) * G4_ + col);
      const unsigned ldso = SGI_OFF1 + (unsigned)flat4 * 4u;
      asm volatile("global_load_async_to_lds_b128 %0, %1, off"
                   :: "v"(ldso), "v"(ga) : "memory");
    }

    // ---- gh = h @ whh^T + b_hh (pre-LN), staged to LDS
    v8f acc[8];
#pragma unroll
    for (int nt = 0; nt < 8; ++nt) acc[nt] = z8;
    for (int kt = 0; kt < H_; kt += 32) {
      v16h a = ldfrag(sh + kt, H_);                 // from LDS
#pragma unroll
      for (int nt = 0; nt < 8; ++nt) {
        v16h bf = ldfrag(whh + (size_t)(nSlab + nt * 16) * H_ + kt, H_);
        acc[nt] = wmma32(a, bf, acc[nt]);
      }
    }
    {
      const int nOff = lane & 15, mTop = (lane >> 4) << 3;
#pragma unroll
      for (int nt = 0; nt < 8; ++nt) {
        const int n = nSlab + nt * 16 + nOff;
        const float bv = b_hh[n];
#pragma unroll
        for (int v = 0; v < 8; ++v)
          sgh[(mTop + v) * G4_ + n] = acc[nt][v] + bv;
      }
    }
    asm volatile("s_wait_asynccnt 0x0" ::: "memory");   // gi tile landed in LDS
    __syncthreads();

    // LayerNorm stats over 4H, wave per row
    {
      const int r = wid;
      float s = 0.f, s2 = 0.f;
      for (int i = lane; i < G4_; i += 32) { float v = sgh[r * G4_ + i]; s += v; s2 += v * v; }
#pragma unroll
      for (int o = 16; o; o >>= 1) { s += __shfl_xor(s, o, 32); s2 += __shfl_xor(s2, o, 32); }
      if (lane == 0) {
        const float m = s / G4_;
        smean[r] = m;
        srstd[r] = rsqrtf(s2 / G4_ - m * m + EPS_);
      }
    }
    __syncthreads();

    // gates i/f/g -> c update (gi tile read from LDS)
    for (int p = 0; p < 16; ++p) {
      const int idx = tid + p * 512;
      const int r = idx >> 9, j = idx & (H_ - 1);
      const float m = smean[r], rs = srstd[r];
      const float zi = sgi[r * G4_ + j]          + (sgh[r * G4_ + j]          - m) * rs * g_hh[j]          + be_hh[j];
      const float zf = sgi[r * G4_ + j + H_]     + (sgh[r * G4_ + j + H_]     - m) * rs * g_hh[j + H_]     + be_hh[j + H_];
      const float zg = sgi[r * G4_ + j + 2 * H_] + (sgh[r * G4_ + j + 2 * H_] - m) * rs * g_hh[j + 2 * H_] + be_hh[j + 2 * H_];
      sc[r * H_ + j] = sigm(zf) * sc[r * H_ + j] + sigm(zi) * tanhf(zg);
    }
    __syncthreads();

    // LayerNorm stats over c
    {
      const int r = wid;
      float s = 0.f, s2 = 0.f;
      for (int i = lane; i < H_; i += 32) { float v = sc[r * H_ + i]; s += v; s2 += v * v; }
#pragma unroll
      for (int o = 16; o; o >>= 1) { s += __shfl_xor(s, o, 32); s2 += __shfl_xor(s2, o, 32); }
      if (lane == 0) {
        const float m = s / H_;
        scm[r] = m;
        scr[r] = rsqrtf(s2 / H_ - m * m + EPS_);
      }
    }
    __syncthreads();

    // o gate -> h
    for (int p = 0; p < 16; ++p) {
      const int idx = tid + p * 512;
      const int r = idx >> 9, j = idx & (H_ - 1);
      const float zo = sgi[r * G4_ + j + 3 * H_] +
                       (sgh[r * G4_ + j + 3 * H_] - smean[r]) * srstd[r] * g_hh[j + 3 * H_] + be_hh[j + 3 * H_];
      const float cn = (sc[r * H_ + j] - scm[r]) * scr[r] * g_c[j] + be_c[j];
      const float h = sigm(zo) * tanhf(cn);
      sh[r * H_ + j] = (h16)h;
      outh[((size_t)(b0 + r) * T_ + t) * H_ + j] = (h16)h;
    }
    __syncthreads();
  }
}

// ------------------------------------------- layer 2: Peephole LSTM scan ----
// Same structure; the gi(t) tile is DMA'd by the Tensor Data Mover.
// LDS: sh(16384) sc(32768) sgh(131072) sgi(131072)
#define SGI_OFF2 180224u
__global__ __launch_bounds__(512)
void rec_peephole_kernel(const float* __restrict__ gi,   // [B,T,4H]
                         const h16* __restrict__ whh,    // [4H,H] f16
                         const float* __restrict__ b_hh,
                         const float* __restrict__ wci, const float* __restrict__ wcf,
                         const float* __restrict__ wco,
                         h16* __restrict__ outh) {       // [B,T,H] f16
  extern __shared__ unsigned char smem[];
  h16*   sh  = (h16*)smem;
  float* sc  = (float*)(smem + (size_t)16 * H_ * sizeof(h16));
  float* sgh = sc + 16 * H_;
  float* sgi = (float*)(smem + SGI_OFF2);

  const int b0 = blockIdx.x * 16;
  const int tid = threadIdx.x, wid = tid >> 5, lane = tid & 31;
  const int nSlab = wid * 128;
  const v8f z8 = {0.f, 0.f, 0.f, 0.f, 0.f, 0.f, 0.f, 0.f};

  for (int i = tid; i < 16 * H_; i += 512) { sh[i] = (h16)0.f; sc[i] = 0.f; }
  __syncthreads();

  for (int t = 0; t < T_; ++t) {
    // ---- TDM: one wave DMAs the 16x2048 f32 gi(t) tile into LDS.
    // 2D tile: tile_dim0=2048 (row), tile_dim1=16 (batch rows),
    // tensor_dim0_stride = T*4H elements (stride between batch rows).
    if (wid == 0) {
      const unsigned long long ga = (unsigned long long)(uintptr_t)
          (gi + ((size_t)b0 * T_ + t) * G4_);
      u32x4 g0;
      g0[0] = 1u;                                        // count=1 valid descriptor
      g0[1] = SGI_OFF2;                                  // lds_addr (bytes)
      g0[2] = (unsigned)(ga & 0xFFFFFFFFull);            // global_addr[31:0]
      g0[3] = (unsigned)((ga >> 32) & 0x1FFFFFFull) | (2u << 30);  // addr hi | type=2
      i32x8 g1;
      g1[0] = (int)(2u << 16);                           // data_size=4B, no mcast
      g1[1] = (int)(((unsigned)G4_ & 0xFFFFu) << 16);    // tensor_dim0 lo16
      g1[2] = (int)(16u << 16);                          // tensor_dim0 hi | tensor_dim1=16 lo
      g1[3] = (int)((unsigned)G4_ << 16);                // tensor_dim1 hi | tile_dim0=2048
      g1[4] = 16;                                        // tile_dim1=16, tile_dim2=0
      g1[5] = (int)(T_ * G4_);                           // tensor_dim0_stride lo32
      g1[6] = 0;                                         // stride0 hi | stride1 lo
      g1[7] = 0;
      i32x4 gz = {0, 0, 0, 0};
#if defined(__clang_major__) && (__clang_major__ >= 23)
      i32x8 gz8 = {0, 0, 0, 0, 0, 0, 0, 0};
      __builtin_amdgcn_tensor_load_to_lds(g0, g1, gz, gz, gz8, 0);
#else
      __builtin_amdgcn_tensor_load_to_lds(g0, g1, gz, gz, 0);
#endif
    }

    v8f acc[8];
#pragma unroll
    for (int nt = 0; nt < 8; ++nt) acc[nt] = z8;
    for (int kt = 0; kt < H_; kt += 32) {
      v16h a = ldfrag(sh + kt, H_);
#pragma unroll
      for (int nt = 0; nt < 8; ++nt) {
        v16h bf = ldfrag(whh + (size_t)(nSlab + nt * 16) * H_ + kt, H_);
        acc[nt] = wmma32(a, bf, acc[nt]);
      }
    }
    {
      const int nOff = lane & 15, mTop = (lane >> 4) << 3;
#pragma unroll
      for (int nt = 0; nt < 8; ++nt) {
        const int n = nSlab + nt * 16 + nOff;
        const float bv = b_hh[n];
#pragma unroll
        for (int v = 0; v < 8; ++v)
          sgh[(mTop + v) * G4_ + n] = acc[nt][v] + bv;
      }
    }
    if (wid == 0) __builtin_amdgcn_s_wait_tensorcnt(0);  // TDM tile landed
    __syncthreads();

    for (int p = 0; p < 16; ++p) {
      const int idx = tid + p * 512;
      const int r = idx >> 9, j = idx & (H_ - 1);
      const float c0 = sc[r * H_ + j];
      const float zi = sgi[r * G4_ + j]          + sgh[r * G4_ + j];
      const float zf = sgi[r * G4_ + j + H_]     + sgh[r * G4_ + j + H_];
      const float zg = sgi[r * G4_ + j + 2 * H_] + sgh[r * G4_ + j + 2 * H_];
      const float zo = sgi[r * G4_ + j + 3 * H_] + sgh[r * G4_ + j + 3 * H_];
      const float ig = sigm(zi + wci[j] * c0);
      const float fg = sigm(zf + wcf[j] * c0);
      const float c  = fg * c0 + ig * tanhf(zg);
      const float og = sigm(zo + wco[j] * c);
      const float h  = og * tanhf(c);
      sc[r * H_ + j] = c;
      sh[r * H_ + j] = (h16)h;
      outh[((size_t)(b0 + r) * T_ + t) * H_ + j] = (h16)h;
    }
    __syncthreads();
  }
}

// -------------------------------------------------- layer 3: IndRNN scan ----
__global__ __launch_bounds__(256)
void rec_indrnn_kernel(const float* __restrict__ gi,  // [B,T,H]
                       const float* __restrict__ whh, // [H]
                       float* __restrict__ out) {     // [B,T,H]
  const int idx = blockIdx.x * 256 + threadIdx.x;     // 0 .. B*H-1
  const int b = idx / H_, j = idx - (idx / H_) * H_;
  const float w = whh[j];
  float h = 0.f;
  for (int t = 0; t < T_; ++t) {
    const size_t o = ((size_t)b * T_ + t) * H_ + j;
    const float v = gi[o] + h * w;
    h = v > 0.f ? v : 0.f;
    out[o] = h;
  }
}

// ------------------------------------------------------------------ launch ---
extern "C" void kernel_launch(void* const* d_in, const int* in_sizes, int n_in,
                              void* d_out, int out_size, void* d_ws, size_t ws_size,
                              hipStream_t stream) {
  const float* x      = (const float*)d_in[0];
  const float* w_ih1  = (const float*)d_in[1];
  const float* w_hh1  = (const float*)d_in[2];
  const float* b_ih1  = (const float*)d_in[3];
  const float* b_hh1  = (const float*)d_in[4];
  const float* g_ih1  = (const float*)d_in[5];
  const float* be_ih1 = (const float*)d_in[6];
  const float* g_hh1  = (const float*)d_in[7];
  const float* be_hh1 = (const float*)d_in[8];
  const float* g_c1   = (const float*)d_in[9];
  const float* be_c1  = (const float*)d_in[10];
  const float* w_ih2  = (const float*)d_in[11];
  const float* w_hh2  = (const float*)d_in[12];
  const float* w_ci2  = (const float*)d_in[13];
  const float* w_cf2  = (const float*)d_in[14];
  const float* w_co2  = (const float*)d_in[15];
  const float* b_ih2  = (const float*)d_in[16];
  const float* b_hh2  = (const float*)d_in[17];
  const float* w_ih3  = (const float*)d_in[18];
  const float* w_hh3  = (const float*)d_in[19];
  const float* b_ih3  = (const float*)d_in[20];
  (void)in_sizes; (void)n_in; (void)out_size; (void)ws_size;

  uint8_t* ws = (uint8_t*)d_ws;
  size_t off = 0;
  auto alloc = [&](size_t bytes) -> void* {
    void* p = ws + off;
    off = (off + bytes + 255) & ~(size_t)255;
    return p;
  };
  float* gi   = (float*)alloc((size_t)B_ * T_ * G4_ * 4);  // reused gi1/gi2/gi3
  h16* xh     = (h16*)alloc((size_t)B_ * T_ * I_ * 2);
  h16* wih1h  = (h16*)alloc((size_t)G4_ * I_ * 2);
  h16* whh1h  = (h16*)alloc((size_t)G4_ * H_ * 2);
  h16* wih2h  = (h16*)alloc((size_t)G4_ * H_ * 2);
  h16* whh2h  = (h16*)alloc((size_t)G4_ * H_ * 2);
  h16* wih3h  = (h16*)alloc((size_t)H_ * H_ * 2);
  h16* out1h  = (h16*)alloc((size_t)B_ * T_ * H_ * 2);
  h16* out2h  = (h16*)alloc((size_t)B_ * T_ * H_ * 2);

  auto cvt = [&](const float* src, h16* dst, size_t n) {
    int blocks = (int)((n + 255) / 256);
    if (blocks > 4096) blocks = 4096;
    f32_to_f16_kernel<<<blocks, 256, 0, stream>>>(src, dst, n);
  };
  cvt(x,     xh,    (size_t)B_ * T_ * I_);
  cvt(w_ih1, wih1h, (size_t)G4_ * I_);
  cvt(w_hh1, whh1h, (size_t)G4_ * H_);
  cvt(w_ih2, wih2h, (size_t)G4_ * H_);
  cvt(w_hh2, whh2h, (size_t)G4_ * H_);
  cvt(w_ih3, wih3h, (size_t)H_ * H_);

  const int MROWS = B_ * T_;  // 32768
  const size_t lds1 = (size_t)SGI_OFF1 + (size_t)16 * G4_ * 4;  // ~311.5 KB
  const size_t lds2 = (size_t)SGI_OFF2 + (size_t)16 * G4_ * 4;  // ~311.3 KB

  // Layer 1
  gemm_bias_f16<<<dim3(G4_ / 128, MROWS / 128), 256, 0, stream>>>(
      xh, wih1h, b_ih1, gi, MROWS, G4_, I_);
  ln_rows_kernel<<<MROWS, 256, 0, stream>>>(gi, g_ih1, be_ih1, G4_);
  rec_lnlstm_kernel<<<B_ / 16, 512, lds1, stream>>>(
      gi, whh1h, b_hh1, g_hh1, be_hh1, g_c1, be_c1, out1h);

  // Layer 2
  gemm_bias_f16<<<dim3(G4_ / 128, MROWS / 128), 256, 0, stream>>>(
      out1h, wih2h, b_ih2, gi, MROWS, G4_, H_);
  rec_peephole_kernel<<<B_ / 16, 512, lds2, stream>>>(
      gi, whh2h, b_hh2, w_ci2, w_cf2, w_co2, out2h);

  // Layer 3
  gemm_bias_f16<<<dim3(H_ / 128, MROWS / 128), 256, 0, stream>>>(
      out2h, wih3h, b_ih3, gi, MROWS, H_, H_);
  rec_indrnn_kernel<<<(B_ * H_) / 256, 256, 0, stream>>>(gi, w_hh3, (float*)d_out);
}
